// PatchGraphAttention_15178414424120
// MI455X (gfx1250) — matrix operand order repaired and checked by reading, not compiled, not run on hardware
//
#include <hip/hip_runtime.h>
#include <hip/hip_bf16.h>
#include <math.h>

#define D_MODEL 1024
#define NHEAD   16
#define D_HEAD  64
#define BATCH   4
#define SEQ     2048
#define M_TOK   (BATCH*SEQ)   // 8192

typedef __attribute__((ext_vector_type(16))) __bf16 v16bf;
typedef __attribute__((ext_vector_type(8)))  __bf16 v8bf;
typedef __attribute__((ext_vector_type(8)))  float  v8f;

union ABFrag { v16bf v; v8bf h[2]; };

// Load a fragment whose 16 bf16 values live as two 8-element (16-byte) chunks.
static __device__ __forceinline__ v16bf load_frag_split(const __bf16* lo, const __bf16* hi) {
  ABFrag u;
  u.h[0] = *reinterpret_cast<const v8bf*>(lo);
  u.h[1] = *reinterpret_cast<const v8bf*>(hi);
  return u.v;
}

// CDNA5 async copy: global -> LDS, 16 bytes per lane, tracked by ASYNCcnt.
static __device__ __forceinline__ void async_g2l_b128(unsigned lds_off, const __bf16* g) {
  unsigned long long ga = (unsigned long long)(size_t)g;
  asm volatile("global_load_async_to_lds_b128 %0, %1, off"
               :: "v"(lds_off), "v"(ga) : "memory");
}

// ---------------------------------------------------------------------------
// Prep: fp32 -> bf16 casts, time-bias table, mask-additive table.
// tab[d] = log(exp(-d) + 1e-12)  (dist/TAU == |i-j| because DELTA == TAU == 300)
// maskadd[b*P+k] = (patch_mask<=0) ? -1e9 : 0
// ---------------------------------------------------------------------------
__global__ void prep_kernel(const float* __restrict__ x,  const float* __restrict__ pm,
                            const float* __restrict__ Wq, const float* __restrict__ Wk,
                            const float* __restrict__ Wv, const float* __restrict__ Wo,
                            __bf16* __restrict__ xb, __bf16* __restrict__ wb,
                            float* __restrict__ tab, float* __restrict__ maskadd) {
  const long long NX = (long long)M_TOK * D_MODEL;
  const long long NW = (long long)D_MODEL * D_MODEL;
  const long long total = NX + 4*NW + SEQ + M_TOK;
  for (long long i = (long long)blockIdx.x*blockDim.x + threadIdx.x; i < total;
       i += (long long)gridDim.x*blockDim.x) {
    if (i < NX) {
      xb[i] = (__bf16)x[i];
    } else if (i < NX + 4*NW) {
      long long j = i - NX;
      int w = (int)(j / NW);
      long long e = j % NW;
      const float* src = (w==0) ? Wq : (w==1) ? Wk : (w==2) ? Wv : Wo;
      wb[j] = (__bf16)src[e];
    } else if (i < NX + 4*NW + SEQ) {
      long long t = i - NX - 4*NW;
      float d = (float)t;
      tab[t] = __logf(__expf(-d) + 1e-12f);
    } else {
      long long t = i - NX - 4*NW - SEQ;
      maskadd[t] = (pm[t] > 0.0f) ? 0.0f : -1e9f;
    }
  }
}

// ---------------------------------------------------------------------------
// GEMM: Y[m,n] = sum_k X[m,k] * W[n,k] + bias[n]
// X: [M,1024] bf16 row-major; W: [1024,1024] bf16 row-major (A*B^T WMMA layout).
// Wave computes 64x64 via 4x4 v_wmma_f32_16x16x32_bf16; WG = 8 waves (2M x 4N)
// tiles 128(M) x 256(N).
// B (weight) tiles are double-buffer staged into LDS with
// global_load_async_to_lds_b128 (ASYNCcnt), so the 256 shared weight columns
// are fetched once per WG per K-step and the copy overlaps the WMMAs.
// mode 0: store bf16 into [b,h,p,dh]          (Q / K)
// mode 2: store bf16 into [b,h,dh,p]          (V transposed for attention)
// mode 3: store fp32 (y + bias) * patch_mask  (final output)
// ---------------------------------------------------------------------------
#define LROW 40                      // 32 data + 8 pad bf16 elems (80B, 16B-aligned rows)
#define LBUF (256*LROW)              // elems per stage buffer

__global__ void __launch_bounds__(256)
gemm_wmma(const __bf16* __restrict__ X, const __bf16* __restrict__ W,
          const float* __restrict__ bias, const float* __restrict__ pmask,
          void* __restrict__ outp, int mode, int gridN) {
  __shared__ __align__(16) __bf16 lbuf[2*LBUF];   // 40 KB

  const int tid  = threadIdx.x;
  const int blk  = blockIdx.x;
  const int blkN = blk % gridN;
  const int blkM = blk / gridN;
  const int wid  = tid >> 5;
  const int lane = tid & 31;
  const int ln   = lane & 15;
  const int hi   = lane >> 4;
  const int mbase = blkM*128 + (wid >> 2)*64;
  const int nbase = blkN*256 + (wid &  3)*64;
  const int nwg   = blkN*256;

  // This thread's 4 copy chunks: chunk id c = i*256+tid -> row = c>>2, ko = (c&3)*8
  const __bf16* wrow[4];
  unsigned      lrow[4];
  #pragma unroll
  for (int i = 0; i < 4; ++i) {
    const int c   = i*256 + tid;
    const int row = c >> 2;
    const int ko  = (c & 3) * 8;
    wrow[i] = W + (size_t)(nwg + row)*D_MODEL + ko;          // + kk at issue time
    lrow[i] = (unsigned)(size_t)(lbuf + row*LROW + ko);      // + buf offset at issue
  }

  // Prologue: stage 0 -> buffer 0
  #pragma unroll
  for (int i = 0; i < 4; ++i) async_g2l_b128(lrow[i], wrow[i]);

  v8f c[4][4] = {};

  for (int kk = 0; kk < D_MODEL; kk += 32) {
    const int cur = (kk >> 5) & 1;
    // Issue next stage into the other buffer, then wait for current stage.
    if (kk + 32 < D_MODEL) {
      #pragma unroll
      for (int i = 0; i < 4; ++i)
        async_g2l_b128(lrow[i] + (1 - cur)*(LBUF*2), wrow[i] + kk + 32);
      asm volatile("s_wait_asynccnt 4" ::: "memory");   // current stage landed
    } else {
      asm volatile("s_wait_asynccnt 0" ::: "memory");
    }
    __syncthreads();                                    // stage visible WG-wide

    v16bf a[4], b[4];
#pragma unroll
    for (int im = 0; im < 4; ++im) {
      // A 16x32 layout: lane<16 holds k = 0..15, lane>=16 -> k = 16..31
      const __bf16* base = X + (size_t)(mbase + im*16 + ln)*D_MODEL + kk + hi*16;
      a[im] = load_frag_split(base, base + 8);
    }
#pragma unroll
    for (int jn = 0; jn < 4; ++jn) {
      // B 32x16 layout from LDS: lane<16 -> k=0..15 of col n, lane>=16 -> k=16..31
      const __bf16* base = lbuf + cur*LBUF + ((wid&3)*64 + jn*16 + ln)*LROW + hi*16;
      b[jn] = load_frag_split(base, base + 8);
    }
#pragma unroll
    for (int im = 0; im < 4; ++im)
#pragma unroll
      for (int jn = 0; jn < 4; ++jn)
        c[im][jn] = __builtin_amdgcn_wmma_f32_16x16x32_bf16(
            false, a[im], false, b[jn], (short)0, c[im][jn], false, false);

    __syncthreads();                                    // protect buffer reuse
  }

  // Epilogue: C/D layout -> element (m = r + 8*hi, n = ln) per fragment
#pragma unroll
  for (int im = 0; im < 4; ++im) {
#pragma unroll
    for (int jn = 0; jn < 4; ++jn) {
      const int n = nbase + jn*16 + ln;
      const float bv = bias ? bias[n] : 0.0f;
#pragma unroll
      for (int r = 0; r < 8; ++r) {
        const int m = mbase + im*16 + r + 8*hi;
        const float y = c[im][jn][r] + bv;
        if (mode == 0) {            // [b,h,p,dh] bf16
          const int bb = m >> 11, p = m & (SEQ-1);
          const int h = n >> 6, dh = n & 63;
          ((__bf16*)outp)[(((size_t)(bb*NHEAD + h)*SEQ + p) << 6) + dh] = (__bf16)y;
        } else if (mode == 2) {     // [b,h,dh,p] bf16 (V transposed)
          const int bb = m >> 11, p = m & (SEQ-1);
          const int h = n >> 6, dh = n & 63;
          ((__bf16*)outp)[(((size_t)(bb*NHEAD + h)*D_HEAD + dh) << 11) + p] = (__bf16)y;
        } else {                    // final fp32 with output mask
          ((float*)outp)[(size_t)m*D_MODEL + n] = y * pmask[m];
        }
      }
    }
  }
}

// ---------------------------------------------------------------------------
// Fused flash attention (online softmax), one wave = 16 query rows of one (b,h).
// Computes S^T = K * Q^T and O^T = V^T * P^T so the query index sits on the
// lane axis: softmax reductions are in-lane + one shfl_xor(16), and the online
// rescale of the O accumulator is a single per-lane scalar multiply.
// ---------------------------------------------------------------------------
__global__ void __launch_bounds__(128)
flash_attn(const __bf16* __restrict__ Qb, const __bf16* __restrict__ Kb,
           const __bf16* __restrict__ Vt, const float* __restrict__ tab,
           const float* __restrict__ maskadd, __bf16* __restrict__ Ob) {
  const float scale = 0.125f; // 1/sqrt(64)
  const int wid  = threadIdx.x >> 5;
  const int lane = threadIdx.x & 31;
  const int ln   = lane & 15;
  const int hi   = lane >> 4;
  const int qblocks = SEQ / 64;
  const int bh   = blockIdx.x / qblocks;
  const int qblk = blockIdx.x % qblocks;
  const int b = bh / NHEAD, h = bh % NHEAD;
  const int qbase = qblk*64 + wid*16;
  const int qg = qbase + ln;   // this lane's query row

  // Q as B-operand of S^T = K*Q^T: lane ln holds column (query) qg,
  // fragment f covers d = 32f + hi*16 + 0..15 (contiguous 32B).
  const __bf16* qptr = Qb + ((size_t)bh*SEQ + qg)*D_HEAD;
  const v16bf Qf0 = *reinterpret_cast<const v16bf*>(qptr + hi*16);
  const v16bf Qf1 = *reinterpret_cast<const v16bf*>(qptr + 32 + hi*16);

  v8f acc[4] = {};               // O^T accumulators (4 dh-tiles of 16)
  float rowmax = -INFINITY, l = 0.0f;
  const float*  madd = maskadd + b*SEQ;
  const __bf16* kbh  = Kb + (size_t)bh*SEQ*D_HEAD;
  const __bf16* vbh  = Vt + (size_t)bh*D_HEAD*SEQ;

  for (int cb = 0; cb < SEQ; cb += 32) {        // 32-key chunks
    // --- S^T tiles: A = K rows (16 keys x 64d), B = Q^T ---
    v8f s[2];
#pragma unroll
    for (int t = 0; t < 2; ++t) {
      const __bf16* kb = kbh + (size_t)(cb + t*16 + ln)*D_HEAD + hi*16;
      v16bf ka0 = load_frag_split(kb,      kb + 8);    // d 0..31
      v16bf ka1 = load_frag_split(kb + 32, kb + 40);   // d 32..63
      v8f z = {};
      z    = __builtin_amdgcn_wmma_f32_16x16x32_bf16(false, ka0, false, Qf0, (short)0, z, false, false);
      s[t] = __builtin_amdgcn_wmma_f32_16x16x32_bf16(false, ka1, false, Qf1, (short)0, z, false, false);
    }

    // Scale + time-bias + key-padding mask. Element (t,r): key = cb+16t+r+8hi.
    float val[2][8];
#pragma unroll
    for (int t = 0; t < 2; ++t)
#pragma unroll
      for (int r = 0; r < 8; ++r) {
        const int kg = cb + t*16 + r + 8*hi;
        int dd = qg - kg; dd = dd < 0 ? -dd : dd;
        val[t][r] = s[t][r]*scale + tab[dd] + madd[kg];
      }

    // Online softmax: per-query (= per-lane) reduction over 16 vals + partner half
    float tmax = val[0][0];
#pragma unroll
    for (int t = 0; t < 2; ++t)
#pragma unroll
      for (int r = 0; r < 8; ++r) tmax = fmaxf(tmax, val[t][r]);
    tmax = fmaxf(tmax, __shfl_xor(tmax, 16));
    const float nm = fmaxf(rowmax, tmax);
    const float corr = __expf(rowmax - nm);
    rowmax = nm;

    float p[2][8], ts = 0.0f;
#pragma unroll
    for (int t = 0; t < 2; ++t)
#pragma unroll
      for (int r = 0; r < 8; ++r) { p[t][r] = __expf(val[t][r] - nm); ts += p[t][r]; }
    ts += __shfl_xor(ts, 16);
    l = l*corr + ts;
#pragma unroll
    for (int tt = 0; tt < 4; ++tt)
#pragma unroll
      for (int r = 0; r < 8; ++r) acc[tt][r] *= corr;

    // Build P^T as B-operand (k = key 0..31, n = query): half-wave exchange.
    ABFrag pb;
#pragma unroll
    for (int r = 0; r < 8; ++r) {
      const float o0 = __shfl_xor(p[0][r], 16);
      const float o1 = __shfl_xor(p[1][r], 16);
      const float f0 = hi ? o1      : p[0][r];  // element r   -> k = 16*hi + r
      const float f1 = hi ? p[1][r] : o0;       // element 8+r -> k = 16*hi + 8 + r
      pb.v[r]     = (__bf16)f0;
      pb.v[8 + r] = (__bf16)f1;
    }

    // O^T += V^T * P^T : A = V^T (dh x key), contiguous because V is stored [dh][p]
#pragma unroll
    for (int tt = 0; tt < 4; ++tt) {
      const __bf16* vb = vbh + (size_t)(tt*16 + ln)*SEQ + cb + hi*16;
      v16bf va = load_frag_split(vb, vb + 8);
      acc[tt] = __builtin_amdgcn_wmma_f32_16x16x32_bf16(false, va, false, pb.v, (short)0, acc[tt], false, false);
    }
  }

  // Normalize and store O as [token, h*64+dh] bf16 for the output GEMM.
  const float rl = 1.0f / l;
  const size_t row = (size_t)b*SEQ + qbase + ln;
#pragma unroll
  for (int tt = 0; tt < 4; ++tt)
#pragma unroll
    for (int r = 0; r < 8; ++r) {
      const int ch = h*64 + tt*16 + r + 8*hi;
      Ob[row*D_MODEL + ch] = (__bf16)(acc[tt][r]*rl);
    }
}

// ---------------------------------------------------------------------------
extern "C" void kernel_launch(void* const* d_in, const int* in_sizes, int n_in,
                              void* d_out, int out_size, void* d_ws, size_t ws_size,
                              hipStream_t stream) {
  const float* x  = (const float*)d_in[0];
  const float* pm = (const float*)d_in[1];
  const float* Wq = (const float*)d_in[2];
  const float* bq = (const float*)d_in[3];
  const float* Wk = (const float*)d_in[4];
  const float* bk = (const float*)d_in[5];
  const float* Wv = (const float*)d_in[6];
  const float* bv = (const float*)d_in[7];
  const float* Wo = (const float*)d_in[8];
  const float* bo = (const float*)d_in[9];

  char* ws = (char*)d_ws;
  const size_t NX = (size_t)M_TOK * D_MODEL;     // 8,388,608 elements
  const size_t NW = (size_t)D_MODEL * D_MODEL;   // 1,048,576 per weight

  // Workspace layout (bytes): xb 16MB | wb 8MB | Qb 16MB | Kb 16MB | Vt 16MB | tab/mask
  // Ob (attention output, bf16) aliases xb: xb is dead after the V projection.
  __bf16* xb  = (__bf16*)ws;
  __bf16* wb  = (__bf16*)(ws + NX*2);
  __bf16* Qb  = (__bf16*)(ws + NX*2 + 4*NW*2);
  __bf16* Kb  = Qb + NX;
  __bf16* Vtb = Kb + NX;
  float*  tab  = (float*)(Vtb + NX);
  float*  madd = tab + SEQ;
  __bf16* Ob  = xb;   // alias (sequential stream ordering makes this safe)

  prep_kernel<<<4096, 256, 0, stream>>>(x, pm, Wq, Wk, Wv, Wo, xb, wb, tab, madd);

  const int gm = M_TOK / 128, gn = D_MODEL / 256;
  gemm_wmma<<<gm*gn, 256, 0, stream>>>(xb, wb + 0*NW, bq, nullptr, Qb,  0, gn);
  gemm_wmma<<<gm*gn, 256, 0, stream>>>(xb, wb + 1*NW, bk, nullptr, Kb,  0, gn);
  gemm_wmma<<<gm*gn, 256, 0, stream>>>(xb, wb + 2*NW, bv, nullptr, Vtb, 2, gn);

  flash_attn<<<BATCH*NHEAD*(SEQ/64), 128, 0, stream>>>(Qb, Kb, Vtb, tab, madd, Ob);

  gemm_wmma<<<gm*gn, 256, 0, stream>>>(Ob, wb + 3*NW, bo, pm, d_out, 3, gn);
}
